// Generator_59906203845176
// MI455X (gfx1250) — matrix operand order repaired and checked by reading, not compiled, not run on hardware
//
#include <hip/hip_runtime.h>

// ---------------------------------------------------------------------------
// Persistent 3-layer LSTM + MLP-head flow generator for MI455X (gfx1250).
// One kernel launch runs the entire flow_length recurrence.
//  - 64 workgroups x 256 threads (8 wave32 waves), batch tile = 32 samples.
//  - All weights bf16-packed into LDS once (WMMA B-fragment layout).
//  - Gates via v_wmma_f32_16x16x32_bf16 chains, f32 accumulators.
//  - Cell state c lives in VGPRs; h exchanged through LDS (bf16, A-frag reads).
// ---------------------------------------------------------------------------

#define FLOW_NUM 2048
#define H 64
#define BT 32                    // batch rows per workgroup
#define NWG (FLOW_NUM / BT)      // 64 workgroups
#define NTHREADS 256

typedef __attribute__((ext_vector_type(16))) __bf16   v16bf;
typedef __attribute__((ext_vector_type(8)))  float    v8f;
typedef __attribute__((ext_vector_type(4)))  unsigned u32x4;

// ----- dynamic LDS budget (bytes) ------------------------------------------
// packed weights: 3x Whh (8192 dw) + 2x Wih (8192 dw) + Wih0 (4096 dw)
// bias 768f, fc weights ~2672f, h_bf 3*32*64 bf16, x_bf 32*32 bf16, y1/y2 f32
constexpr size_t SMEM_BYTES =
    (size_t)(3 * 8192 + 2 * 8192 + 4096) * 4 +          // packed bf16 weights
    (size_t)768 * 4 +                                   // fused biases
    (size_t)(2048 + 32 + 512 + 16 + 48 + 16) * 4 +      // fc weights/biases
    (size_t)(3 * BT * H) * 2 +                          // h_bf per layer
    (size_t)(BT * 32) * 2 +                             // x_bf (padded K=32)
    (size_t)(BT * 32) * 4 +                             // y1
    (size_t)(BT * 16) * 4;                              // y2

// ----- small helpers -------------------------------------------------------
__device__ __forceinline__ unsigned short f2bf(float f) {
    unsigned u = __float_as_uint(f);
    u += 0x7FFFu + ((u >> 16) & 1u);        // round-to-nearest-even
    return (unsigned short)(u >> 16);
}
__device__ __forceinline__ float bf2f(unsigned short h) {
    return __uint_as_float(((unsigned)h) << 16);
}
__device__ __forceinline__ float sigf(float x) { return 1.f / (1.f + __expf(-x)); }
__device__ __forceinline__ float tanh_fast(float x) { return 2.f / (1.f + __expf(-2.f * x)) - 1.f; }
__device__ __forceinline__ float lrelu(float x) { return x > 0.f ? x : 0.01f * x; }

__device__ __forceinline__ v8f splat8(float x) {
    v8f r;
#pragma unroll
    for (int i = 0; i < 8; ++i) r[i] = x;
    return r;
}

__device__ __forceinline__ v8f wmma_bf16(v16bf a, v16bf b, v8f c) {
    // D = A(16x32 bf16) x B(32x16 bf16) + C(16x16 f32)
    return __builtin_amdgcn_wmma_f32_16x16x32_bf16(
        /*neg_a=*/false, a, /*neg_b=*/false, b,
        /*c_mod=*/(short)0, c, /*reuse_a=*/false, /*reuse_b=*/false);
}

// A fragment (16x32 bf16) gathered from LDS row-major h buffer.
// Lane L<16: M=L, K = kbase+{0..7} (v0..3) and kbase+16+{0..7} (v4..7).
// Lane L>=16: same rows, K halves shifted by 8.  -> two ds_load_b128.
__device__ __forceinline__ v16bf load_a_frag(const unsigned short* hbuf, int rowstride,
                                             int rowbase, int kbase, int lane) {
    int m  = rowbase + (lane & 15);
    int kh = (lane >> 4) << 3;              // 0 or 8
    const unsigned short* row = hbuf + m * rowstride;
    union { v16bf v; u32x4 q[2]; } r;
    r.q[0] = *(const u32x4*)(row + kbase + kh);
    r.q[1] = *(const u32x4*)(row + kbase + 16 + kh);
    return r.v;
}

// B fragment (32x16 bf16) from pre-packed LDS: contiguous 8 dwords per lane.
__device__ __forceinline__ v16bf load_b_frag(const unsigned* pk, int ntile, int ktile,
                                             int nkt, int lane) {
    const u32x4* p = (const u32x4*)(pk + ((((ntile * nkt + ktile) << 5) + lane) << 3));
    union { v16bf v; u32x4 q[2]; } r;
    r.q[0] = p[0];
    r.q[1] = p[1];
    return r.v;
}

// Pack a [256][64] f32 weight (gates-major, row n = out col, k = in dim) into
// WMMA-B-fragment order: word[((nt*2+kt)*32+lane)*8+i] = bf16 pair (k, k+1)
// with n = nt*16 + (lane&15), k = kt*32 + (lane>>4)*8 + 2i + (i>=4 ? 8 : 0).
__device__ void pack_weight_256x64(const float* W, unsigned* dst, int tid) {
    for (int w = tid; w < 8192; w += NTHREADS) {
        int i = w & 7, lane = (w >> 3) & 31, kt = (w >> 8) & 1, nt = w >> 9;
        int n = (nt << 4) + (lane & 15);
        int k = (kt << 5) + ((lane >> 4) << 3) + (i << 1) + ((i >> 2) << 3);
        unsigned lo = f2bf(W[n * 64 + k]);
        unsigned hi = f2bf(W[n * 64 + k + 1]);
        dst[w] = lo | (hi << 16);
    }
}

// One LSTM layer for this wave's (rowbase, jgrp) tile.  XK = input width (32
// padded for layer 0, else 64); NKT_IH = K-tiles in packed Wih.
template <int XK, int NKT_IH>
__device__ __forceinline__ void lstm_layer(const unsigned short* xb, int xstride,
                                           const unsigned* pkIH,
                                           unsigned short* hb,        // [BT][64]
                                           const unsigned* pkHH,
                                           const float* biasL,
                                           v8f& c, int rowbase, int jgrp, int lane) {
    __syncthreads();  // new x (h_{l-1}) visible
    v16bf ax0 = load_a_frag(xb, xstride, rowbase, 0, lane);
    v16bf ax1 = (XK == 64) ? load_a_frag(xb, xstride, rowbase, 32, lane) : ax0;
    v16bf ah0 = load_a_frag(hb, H, rowbase, 0, lane);
    v16bf ah1 = load_a_frag(hb, H, rowbase, 32, lane);
    __syncthreads();  // every wave captured old h before we overwrite it

    v8f acc[4];
#pragma unroll
    for (int q = 0; q < 4; ++q) {          // gates: i, f, g, o
        int nt = (q << 2) + jgrp;          // gate-column tile in [0,16)
        v8f a = splat8(biasL[(nt << 4) + (lane & 15)]);
        a = wmma_bf16(ah0, load_b_frag(pkHH, nt, 0, 2, lane), a);
        a = wmma_bf16(ah1, load_b_frag(pkHH, nt, 1, 2, lane), a);
        a = wmma_bf16(ax0, load_b_frag(pkIH, nt, 0, NKT_IH, lane), a);
        if (XK == 64)
            a = wmma_bf16(ax1, load_b_frag(pkIH, nt, 1, NKT_IH, lane), a);
        acc[q] = a;
    }

    // Elementwise cell update in the D-fragment layout:
    // lane&15 -> hidden col j, vgpr e (+8 for lanes>=16) -> row.
    v8f hnew;
#pragma unroll
    for (int e = 0; e < 8; ++e) {
        float iv = sigf(acc[0][e]);
        float fv = sigf(acc[1][e]);
        float gv = tanh_fast(acc[2][e]);
        float ov = sigf(acc[3][e]);
        float cn = fv * c[e] + iv * gv;
        c[e] = cn;
        hnew[e] = ov * tanh_fast(cn);
    }
    int jcol = (jgrp << 4) + (lane & 15);
    int mb   = rowbase + ((lane >> 4) << 3);
#pragma unroll
    for (int e = 0; e < 8; ++e)
        hb[(mb + e) * H + jcol] = f2bf(hnew[e]);
}

__global__ __launch_bounds__(NTHREADS) void lstm_flow_kernel(
    const float* __restrict__ noise,
    const float* __restrict__ w_ih0, const float* __restrict__ w_hh0,
    const float* __restrict__ b_ih0, const float* __restrict__ b_hh0,
    const float* __restrict__ w_ih1, const float* __restrict__ w_hh1,
    const float* __restrict__ b_ih1, const float* __restrict__ b_hh1,
    const float* __restrict__ w_ih2, const float* __restrict__ w_hh2,
    const float* __restrict__ b_ih2, const float* __restrict__ b_hh2,
    const float* __restrict__ fc1_wg, const float* __restrict__ fc1_bg,
    const float* __restrict__ fc2_wg, const float* __restrict__ fc2_bg,
    const float* __restrict__ fc3_wg, const float* __restrict__ fc3_bg,
    const int* __restrict__ flow_len_p,
    float* __restrict__ out) {
    extern __shared__ char smem_raw[];
    char* cur = smem_raw;
    auto carve = [&](size_t bytes) { char* p = cur; cur += bytes; return p; };

    unsigned* pkHH0 = (unsigned*)carve(8192 * 4);
    unsigned* pkHH1 = (unsigned*)carve(8192 * 4);
    unsigned* pkHH2 = (unsigned*)carve(8192 * 4);
    unsigned* pkIH1 = (unsigned*)carve(8192 * 4);
    unsigned* pkIH2 = (unsigned*)carve(8192 * 4);
    unsigned* pkIH0 = (unsigned*)carve(4096 * 4);
    float* bias = (float*)carve(768 * 4);          // fused b_ih+b_hh, 3x256
    float* fc1w = (float*)carve(2048 * 4);
    float* fc1b = (float*)carve(32 * 4);
    float* fc2w = (float*)carve(512 * 4);
    float* fc2b = (float*)carve(16 * 4);
    float* fc3w = (float*)carve(48 * 4);
    float* fc3b = (float*)carve(16 * 4);
    unsigned short* hbf = (unsigned short*)carve(3 * BT * H * 2);
    unsigned short* xbf = (unsigned short*)carve(BT * 32 * 2);
    float* y1 = (float*)carve(BT * 32 * 4);
    float* y2 = (float*)carve(BT * 16 * 4);

    const int tid = threadIdx.x;
    const int lane = tid & 31;
    const int wave = tid >> 5;
    const int rowbase = (wave >> 2) << 4;  // 0 or 16
    const int jgrp = wave & 3;             // 16-wide column group
    const int wg = blockIdx.x;

    // ---------------- prologue: stage everything into LDS ------------------
    pack_weight_256x64(w_hh0, pkHH0, tid);
    pack_weight_256x64(w_hh1, pkHH1, tid);
    pack_weight_256x64(w_hh2, pkHH2, tid);
    pack_weight_256x64(w_ih1, pkIH1, tid);
    pack_weight_256x64(w_ih2, pkIH2, tid);
    for (int w = tid; w < 4096; w += NTHREADS) {   // Wih0: [256][3] -> K=32 pad
        int i = w & 7, ln = (w >> 3) & 31, nt = w >> 8;
        int n = (nt << 4) + (ln & 15);
        int k = ((ln >> 4) << 3) + (i << 1) + ((i >> 2) << 3);
        unsigned lo = (k < 3) ? (unsigned)f2bf(w_ih0[n * 3 + k]) : 0u;
        unsigned hi = (k + 1 < 3) ? (unsigned)f2bf(w_ih0[n * 3 + k + 1]) : 0u;
        pkIH0[w] = lo | (hi << 16);
    }
    for (int i = tid; i < 256; i += NTHREADS) {
        bias[i]       = b_ih0[i] + b_hh0[i];
        bias[256 + i] = b_ih1[i] + b_hh1[i];
        bias[512 + i] = b_ih2[i] + b_hh2[i];
    }
    for (int i = tid; i < 2048; i += NTHREADS) fc1w[i] = fc1_wg[i];
    for (int i = tid; i < 512;  i += NTHREADS) fc2w[i] = fc2_wg[i];
    if (tid < 48) fc3w[tid] = fc3_wg[tid];
    if (tid < 32) fc1b[tid] = fc1_bg[tid];
    if (tid < 16) fc2b[tid] = fc2_bg[tid];
    if (tid < 3)  fc3b[tid] = fc3_bg[tid];
    for (int i = tid; i < 3 * BT * H; i += NTHREADS) hbf[i] = 0;  // h0 = 0
    for (int i = tid; i < BT * 32; i += NTHREADS) {               // x0 = noise
        int r = i >> 5, k = i & 31;
        xbf[i] = (k < 3) ? f2bf(noise[(wg * BT + r) * 3 + k]) : (unsigned short)0;
    }

    const int FL = *flow_len_p;
    v8f c0 = splat8(0.f), c1 = splat8(0.f), c2 = splat8(0.f);

    unsigned short* h0 = hbf;
    unsigned short* h1 = hbf + BT * H;
    unsigned short* h2 = hbf + 2 * BT * H;

    // ---------------- the 500-step recurrence ------------------------------
    for (int t = 0; t < FL; ++t) {
        lstm_layer<32, 1>(xbf, 32, pkIH0, h0, pkHH0, bias,       c0, rowbase, jgrp, lane);
        lstm_layer<64, 2>(h0,  H,  pkIH1, h1, pkHH1, bias + 256, c1, rowbase, jgrp, lane);
        lstm_layer<64, 2>(h1,  H,  pkIH2, h2, pkHH2, bias + 512, c2, rowbase, jgrp, lane);

        // -------- MLP head: 64 -> 32 -> 16 -> 3, LeakyReLU(0.01) -----------
        __syncthreads();
        for (int idx = tid; idx < BT * 32; idx += NTHREADS) {
            int r = idx >> 5, cc = idx & 31;
            float s = fc1b[cc];
            const float* wrow = fc1w + cc * H;
            const unsigned short* hr = h2 + r * H;
#pragma unroll 8
            for (int k = 0; k < H; ++k) s += bf2f(hr[k]) * wrow[k];
            y1[idx] = lrelu(s);
        }
        __syncthreads();
        for (int idx = tid; idx < BT * 16; idx += NTHREADS) {
            int r = idx >> 4, cc = idx & 15;
            float s = fc2b[cc];
            const float* wrow = fc2w + cc * 32;
            const float* yr = y1 + r * 32;
#pragma unroll
            for (int k = 0; k < 32; ++k) s += yr[k] * wrow[k];
            y2[idx] = lrelu(s);
        }
        __syncthreads();
        if (tid < BT * 3) {
            int r = tid / 3, cc = tid % 3;
            float s = fc3b[cc];
            const float* wrow = fc3w + cc * 16;
            const float* yr = y2 + r * 16;
#pragma unroll
            for (int k = 0; k < 16; ++k) s += yr[k] * wrow[k];
            s = lrelu(s);
            out[((size_t)(wg * BT + r) * FL + t) * 3 + cc] = s;  // [N, T, 3]
            xbf[r * 32 + cc] = f2bf(s);                          // next input
        }
        // next iteration's first __syncthreads() publishes xbf
    }
}

extern "C" void kernel_launch(void* const* d_in, const int* in_sizes, int n_in,
                              void* d_out, int out_size, void* d_ws, size_t ws_size,
                              hipStream_t stream) {
    (void)in_sizes; (void)n_in; (void)out_size; (void)d_ws; (void)ws_size;
    const float* noise = (const float*)d_in[0];
    const float* w_ih0 = (const float*)d_in[1];
    const float* w_hh0 = (const float*)d_in[2];
    const float* b_ih0 = (const float*)d_in[3];
    const float* b_hh0 = (const float*)d_in[4];
    const float* w_ih1 = (const float*)d_in[5];
    const float* w_hh1 = (const float*)d_in[6];
    const float* b_ih1 = (const float*)d_in[7];
    const float* b_hh1 = (const float*)d_in[8];
    const float* w_ih2 = (const float*)d_in[9];
    const float* w_hh2 = (const float*)d_in[10];
    const float* b_ih2 = (const float*)d_in[11];
    const float* b_hh2 = (const float*)d_in[12];
    const float* fc1_w = (const float*)d_in[13];
    const float* fc1_b = (const float*)d_in[14];
    const float* fc2_w = (const float*)d_in[15];
    const float* fc2_b = (const float*)d_in[16];
    const float* fc3_w = (const float*)d_in[17];
    const float* fc3_b = (const float*)d_in[18];
    const int*   flowl = (const int*)d_in[19];

    // >64KB dynamic LDS: raise the cap (immediate call, safe under capture).
    (void)hipFuncSetAttribute((const void*)lstm_flow_kernel,
                              hipFuncAttributeMaxDynamicSharedMemorySize,
                              (int)SMEM_BYTES);

    lstm_flow_kernel<<<NWG, NTHREADS, SMEM_BYTES, stream>>>(
        noise, w_ih0, w_hh0, b_ih0, b_hh0,
        w_ih1, w_hh1, b_ih1, b_hh1,
        w_ih2, w_hh2, b_ih2, b_hh2,
        fc1_w, fc1_b, fc2_w, fc2_b, fc3_w, fc3_b,
        flowl, (float*)d_out);
}
